// LMGNN_51977694216650
// MI455X (gfx1250) — matrix operand order repaired and checked by reading, not compiled, not run on hardware
//
#include <hip/hip_runtime.h>
#include <hip/hip_bf16.h>

#define N_USER 100000
#define N_ITEM 150000
#define N_NODE (N_USER + N_ITEM)
#define D 64
#define NNZ 1250000
#define BATCH 16384
#define LSEQ 3
#define GD 16
#define DSTATE 8
#define DINNER 32
#define TEMP 0.8f

typedef __attribute__((ext_vector_type(16))) _Float16 v16h;
typedef __attribute__((ext_vector_type(8)))  float    v8f;

__device__ __forceinline__ float wave_allreduce(float v) {
    // butterfly: every lane ends with the full 32-lane sum
    #pragma unroll
    for (int m = 16; m >= 1; m >>= 1) v += __shfl_xor(v, m, 32);
    return v;
}
__device__ __forceinline__ float siluf(float x)     { return x / (1.0f + __expf(-x)); }
__device__ __forceinline__ float softplusf(float x) { return (x > 20.0f) ? x : log1pf(__expf(x)); }

// ---------------------------------------------------------------------------
// GNN: scatter-add  dst[row] += val * src[col], one thread per (edge, 4 floats)
// ---------------------------------------------------------------------------
__global__ __launch_bounds__(256) void gnn_scatter(
    const int* __restrict__ erow, const int* __restrict__ ecol,
    const float* __restrict__ eval, const float* __restrict__ src,
    float* __restrict__ dst)
{
    long long t = (long long)blockIdx.x * blockDim.x + threadIdx.x;
    if (t >= (long long)NNZ * 16) return;
    int e = (int)(t >> 4);
    int p = (int)(t & 15);
    int col = ecol[e];
    int row = erow[e];
    float v = eval[e];
    const float4 x = *(const float4*)(src + (long long)col * D + p * 4);
    float* dp = dst + (long long)row * D + p * 4;
    atomicAdd(dp + 0, v * x.x);   // non-returning global_atomic_add_f32
    atomicAdd(dp + 1, v * x.y);
    atomicAdd(dp + 2, v * x.z);
    atomicAdd(dp + 3, v * x.w);
}

// one wave per row: L2-normalize 64 floats (2 per lane)
__global__ __launch_bounds__(256) void gnn_normalize(float* __restrict__ E)
{
    int wave = threadIdx.x >> 5;
    int lane = threadIdx.x & 31;
    int row = blockIdx.x * 8 + wave;
    if (row >= N_NODE) return;
    float* p = E + (long long)row * D;
    float a = p[lane], b = p[lane + 32];
    float ss = wave_allreduce(a * a + b * b);
    float inv = 1.0f / fmaxf(sqrtf(ss), 1e-12f);
    p[lane] = a * inv;
    p[lane + 32] = b * inv;
}

// seq[b][l][:] = E[node_ids[b]][:]
__global__ __launch_bounds__(256) void gather_seq(
    const float* __restrict__ E, const int* __restrict__ ids,
    float* __restrict__ seq, int l)
{
    int t = blockIdx.x * blockDim.x + threadIdx.x;
    if (t >= BATCH * D) return;
    int b = t >> 6, k = t & 63;
    seq[((long long)b * LSEQ + l) * D + k] = E[(long long)ids[b] * D + k];
}

// ---------------------------------------------------------------------------
// Down-projection GEMM with WMMA:  g(R x 16) = seq(R x 64) @ down_w.T(64 x 16)
// One wave per 16-row tile, K=64 handled as two 16x16x32 f16 WMMAs.
// A 16-bit 16x32 layout: lane L -> M = L%16, elem i -> K = 16*(i>=8) + 8*(L>=16) + (i%8)
// B 16-bit 32x16 layout: lane L -> N = L%16, elem i -> K = 16*(i>=8) + 8*(L>=16) + (i%8)
// C/D f32 16x16:         lane L -> N = L%16, VGPR r -> M = r + 8*(L>=16)
// ---------------------------------------------------------------------------
__global__ __launch_bounds__(32) void downproj_wmma(
    const float* __restrict__ seq, const float* __restrict__ down_w,
    float* __restrict__ g)
{
    const int tile = blockIdx.x;
    const int lane = threadIdx.x;
    const int m  = lane & 15;
    const int hi = lane >> 4;         // 0 or 1
    const long long row = (long long)tile * 16 + m;

    v8f c = {};
    #pragma unroll
    for (int kc = 0; kc < 64; kc += 32) {
        v16h a, b;
        #pragma unroll
        for (int i = 0; i < 16; ++i) {
            int ka = kc + ((i >= 8) ? 16 : 0) + hi * 8 + (i & 7);
            a[i] = (_Float16)seq[row * D + ka];
            int kb = kc + ((i >= 8) ? 16 : 0) + hi * 8 + (i & 7);
            b[i] = (_Float16)down_w[m * D + kb];  // B[k][n] = down_w[n][k]
        }
        c = __builtin_amdgcn_wmma_f32_16x16x32_f16(
                false, a, false, b, (short)0, c, false, false);
    }
    #pragma unroll
    for (int r = 0; r < 8; ++r) {
        long long om = (long long)tile * 16 + r + hi * 8;
        g[om * GD + m] = c[r];
    }
}

// ---------------------------------------------------------------------------
// Fused Mamba + residual + LayerNorm + logit + softmax + weighted seq fusion.
// One wave32 per batch element; lane = inner channel d (DINNER == 32).
// ---------------------------------------------------------------------------
__global__ __launch_bounds__(32) void mamba_fuse(
    const float* __restrict__ g,          // [BATCH*3*16]
    const float* __restrict__ seq,        // [BATCH*3*64]
    const float* __restrict__ in_proj_w,  // [64*16]
    const float* __restrict__ conv_w,     // [32*4]
    const float* __restrict__ conv_b,     // [32]
    const float* __restrict__ x_proj_w,   // [17*32]
    const float* __restrict__ dt_proj_w,  // [32]
    const float* __restrict__ dt_proj_b,  // [32]
    const float* __restrict__ A_log,      // [32*8]
    const float* __restrict__ D_param,    // [32]
    const float* __restrict__ out_proj_w, // [16*32]
    const float* __restrict__ ln_g, const float* __restrict__ ln_b,
    const float* __restrict__ to_logit_w, // [16]
    const float* __restrict__ to_logit_b, // [1]
    float* __restrict__ out)              // [BATCH*64]
{
    const int b = blockIdx.x;
    const int d = threadIdx.x;            // 0..31  (inner channel)
    const float* gb = g + (long long)b * LSEQ * GD;

    float gl[LSEQ][GD];
    #pragma unroll
    for (int l = 0; l < LSEQ; ++l)
        #pragma unroll
        for (int k = 0; k < GD; ++k) gl[l][k] = gb[l * GD + k];

    // in_proj: x = g @ W[0:32].T, z = g @ W[32:64].T  (per-lane rows)
    float wx[GD], wz[GD];
    #pragma unroll
    for (int k = 0; k < GD; ++k) {
        wx[k] = in_proj_w[d * GD + k];
        wz[k] = in_proj_w[(DINNER + d) * GD + k];
    }
    float x[LSEQ], z[LSEQ];
    #pragma unroll
    for (int l = 0; l < LSEQ; ++l) {
        float ax = 0.0f, az = 0.0f;
        #pragma unroll
        for (int k = 0; k < GD; ++k) { ax += gl[l][k] * wx[k]; az += gl[l][k] * wz[k]; }
        x[l] = ax; z[l] = az;
    }

    // causal depthwise conv (DCONV=4, left-pad 3) + SiLU
    float cw0 = conv_w[d*4+0], cw1 = conv_w[d*4+1], cw2 = conv_w[d*4+2], cw3 = conv_w[d*4+3];
    float cb  = conv_b[d];
    float xc0 = x[0]*cw3 + cb;
    float xc1 = x[0]*cw2 + x[1]*cw3 + cb;
    float xc2 = x[0]*cw1 + x[1]*cw2 + x[2]*cw3 + cb;
    x[0] = siluf(xc0); x[1] = siluf(xc1); x[2] = siluf(xc2);

    // dbc = x @ x_proj_w.T  -> 17 cross-lane contractions per timestep
    float dtr[LSEQ], Bm[LSEQ][DSTATE], Cm[LSEQ][DSTATE];
    #pragma unroll
    for (int l = 0; l < LSEQ; ++l) {
        #pragma unroll
        for (int j = 0; j < 1 + 2 * DSTATE; ++j) {
            float v = wave_allreduce(x[l] * x_proj_w[j * DINNER + d]);
            if (j == 0)                dtr[l] = v;
            else if (j <= DSTATE)      Bm[l][j - 1] = v;
            else                       Cm[l][j - 1 - DSTATE] = v;
        }
    }

    const float dtw = dt_proj_w[d], dtb = dt_proj_b[d];
    float Av[DSTATE];
    #pragma unroll
    for (int s = 0; s < DSTATE; ++s) Av[s] = -__expf(A_log[d * DSTATE + s]);
    const float Dp = D_param[d];

    // selective scan over L=3
    float h[DSTATE];
    #pragma unroll
    for (int s = 0; s < DSTATE; ++s) h[s] = 0.0f;
    float y[LSEQ];
    #pragma unroll
    for (int l = 0; l < LSEQ; ++l) {
        float dt = softplusf(dtr[l] * dtw + dtb);
        float acc = 0.0f;
        #pragma unroll
        for (int s = 0; s < DSTATE; ++s) {
            h[s] = __expf(dt * Av[s]) * h[s] + dt * Bm[l][s] * x[l];
            acc += h[s] * Cm[l][s];
        }
        y[l] = (acc + Dp * x[l]) * siluf(z[l]);
    }

    // out_proj (cross-lane) + residual + LayerNorm + logit
    float logit[LSEQ];
    #pragma unroll
    for (int l = 0; l < LSEQ; ++l) {
        float v[GD];
        float mu = 0.0f;
        #pragma unroll
        for (int j = 0; j < GD; ++j) {
            v[j] = wave_allreduce(y[l] * out_proj_w[j * DINNER + d]) + gl[l][j];
            mu += v[j];
        }
        mu *= (1.0f / GD);
        float var = 0.0f;
        #pragma unroll
        for (int j = 0; j < GD; ++j) { float t = v[j] - mu; var += t * t; }
        var *= (1.0f / GD);
        float inv = rsqrtf(var + 1e-12f);
        float lg = 0.0f;
        #pragma unroll
        for (int j = 0; j < GD; ++j)
            lg += ((v[j] - mu) * inv * ln_g[j] + ln_b[j]) * to_logit_w[j];
        logit[l] = lg + to_logit_b[0];
    }

    // softmax over L with temperature, then fuse seq
    float s0 = logit[0] / TEMP, s1 = logit[1] / TEMP, s2 = logit[2] / TEMP;
    float mx = fmaxf(s0, fmaxf(s1, s2));
    float e0 = __expf(s0 - mx), e1 = __expf(s1 - mx), e2 = __expf(s2 - mx);
    float inv = 1.0f / (e0 + e1 + e2);
    float w0 = e0 * inv, w1 = e1 * inv, w2 = e2 * inv;

    const float* sb = seq + (long long)b * LSEQ * D;
    out[(long long)b * D + d]      = sb[d]        * w0 + sb[D + d]        * w1 + sb[2*D + d]        * w2;
    out[(long long)b * D + 32 + d] = sb[32 + d]   * w0 + sb[D + 32 + d]   * w1 + sb[2*D + 32 + d]   * w2;
}

// ---------------------------------------------------------------------------
extern "C" void kernel_launch(void* const* d_in, const int* in_sizes, int n_in,
                              void* d_out, int out_size, void* d_ws, size_t ws_size,
                              hipStream_t stream) {
    const float* user_emb   = (const float*)d_in[0];
    const float* item_emb   = (const float*)d_in[1];
    const int*   edge_row   = (const int*)d_in[2];
    const int*   edge_col   = (const int*)d_in[3];
    const float* edge_val   = (const float*)d_in[4];
    const int*   node_ids   = (const int*)d_in[5];
    const float* down_w     = (const float*)d_in[6];
    const float* in_proj_w  = (const float*)d_in[7];
    const float* conv_w     = (const float*)d_in[8];
    const float* conv_b     = (const float*)d_in[9];
    const float* x_proj_w   = (const float*)d_in[10];
    const float* dt_proj_w  = (const float*)d_in[11];
    const float* dt_proj_b  = (const float*)d_in[12];
    const float* A_log      = (const float*)d_in[13];
    const float* D_param    = (const float*)d_in[14];
    const float* out_proj_w = (const float*)d_in[15];
    const float* ln_g       = (const float*)d_in[16];
    const float* ln_b       = (const float*)d_in[17];
    const float* to_logit_w = (const float*)d_in[18];
    const float* to_logit_b = (const float*)d_in[19];
    float* outp = (float*)d_out;

    // workspace layout (floats)
    float* EA  = (float*)d_ws;                       // N*64
    float* EB  = EA + (size_t)N_NODE * D;            // N*64
    float* seq = EB + (size_t)N_NODE * D;            // BATCH*3*64
    float* gws = seq + (size_t)BATCH * LSEQ * D;     // BATCH*3*16

    const size_t embBytes = (size_t)N_NODE * D * sizeof(float);

    // embeds = concat(user, item)
    hipMemcpyAsync(EA, user_emb, (size_t)N_USER * D * sizeof(float),
                   hipMemcpyDeviceToDevice, stream);
    hipMemcpyAsync(EA + (size_t)N_USER * D, item_emb, (size_t)N_ITEM * D * sizeof(float),
                   hipMemcpyDeviceToDevice, stream);

    const int gatherBlocks  = (BATCH * D + 255) / 256;
    const int scatterBlocks = (int)(((long long)NNZ * 16 + 255) / 256);
    const int normBlocks    = (N_NODE + 7) / 8;

    gather_seq<<<gatherBlocks, 256, 0, stream>>>(EA, node_ids, seq, 0);

    // GNN layer 1: EA -> EB
    hipMemsetAsync(EB, 0, embBytes, stream);
    gnn_scatter<<<scatterBlocks, 256, 0, stream>>>(edge_row, edge_col, edge_val, EA, EB);
    gnn_normalize<<<normBlocks, 256, 0, stream>>>(EB);
    gather_seq<<<gatherBlocks, 256, 0, stream>>>(EB, node_ids, seq, 1);

    // GNN layer 2: EB -> EA
    hipMemsetAsync(EA, 0, embBytes, stream);
    gnn_scatter<<<scatterBlocks, 256, 0, stream>>>(edge_row, edge_col, edge_val, EB, EA);
    gnn_normalize<<<normBlocks, 256, 0, stream>>>(EA);
    gather_seq<<<gatherBlocks, 256, 0, stream>>>(EA, node_ids, seq, 2);

    // down-projection via WMMA: 49152 rows / 16 per wave
    downproj_wmma<<<(BATCH * LSEQ) / 16, 32, 0, stream>>>(seq, down_w, gws);

    // fused mamba + LN + softmax fusion: one wave per batch row
    mamba_fuse<<<BATCH, 32, 0, stream>>>(
        gws, seq, in_proj_w, conv_w, conv_b, x_proj_w, dt_proj_w, dt_proj_b,
        A_log, D_param, out_proj_w, ln_g, ln_b, to_logit_w, to_logit_b, outp);
}
